// HVIN_7636451852549
// MI455X (gfx1250) — matrix operand order repaired and checked by reading, not compiled, not run on hardware
//
#include <hip/hip_runtime.h>
#include <hip/hip_bf16.h>
#include <math.h>

typedef __attribute__((ext_vector_type(16))) _Float16 v16h;
typedef __attribute__((ext_vector_type(8)))  float    v8f;

// ---------------------------------------------------------------------------
// Custom im2col K-permutation (A and B must agree; GEMM is K-order invariant):
//   hwK 0..7   = plane0 taps 0..7        (laneLo e=0..7)
//   hwK 16     = plane0 tap 8            (laneLo e=8)
//   hwK 8..15  = plane1 taps 0..7        (laneHi e=0..7 via base+PSZ)
//   hwK 24     = plane1 tap 8            (laneHi e=8)
//   CIN==3:  hwK 17..23 = plane2 taps 0..6 (laneLo e=9..15)
//            hwK 25,26  = plane2 taps 7,8  (laneHi e=9,10)
// A-build: 9 (or 18) unconditional ds loads with compile-time offsets from a
// per-lane base; laneHi just adds PSZ. No guards, no div/mod in the hot path.
// ---------------------------------------------------------------------------
template <int CIN, int PSZ, int PW>
__device__ __forceinline__ v16h build_a(const _Float16* __restrict__ PL,
                                        int pixoff, bool hi) {
  const _Float16* bp = PL + (hi ? PSZ : 0) + pixoff;
  v16h a;
  a[0] = bp[0];          a[1] = bp[1];          a[2] = bp[2];
  a[3] = bp[PW];         a[4] = bp[PW + 1];     a[5] = bp[PW + 2];
  a[6] = bp[2 * PW];     a[7] = bp[2 * PW + 1]; a[8] = bp[2 * PW + 2];
  if (CIN == 3) {
    const _Float16* vp = PL + 2 * PSZ + pixoff;
    _Float16 t0 = vp[0],      t1 = vp[1],          t2 = vp[2];
    _Float16 t3 = vp[PW],     t4 = vp[PW + 1],     t5 = vp[PW + 2];
    _Float16 t6 = vp[2 * PW], t7 = vp[2 * PW + 1], t8 = vp[2 * PW + 2];
    const _Float16 z = (_Float16)0.f;
    a[9]  = hi ? t7 : t0;
    a[10] = hi ? t8 : t1;
    a[11] = hi ? z : t2;
    a[12] = hi ? z : t3;
    a[13] = hi ? z : t4;
    a[14] = hi ? z : t5;
    a[15] = hi ? z : t6;
  } else {
    const _Float16 z = (_Float16)0.f;
    a[9] = z; a[10] = z; a[11] = z; a[12] = z; a[13] = z; a[14] = z; a[15] = z;
  }
  return a;
}

// B operand matching the permutation above. w points at 9*CIN weights for one
// (clamped) output channel. Loads are UNCONDITIONAL: garbage weights for
// invalid channels are harmless (their outputs are never consumed), which
// avoids predicated loads / exec-mask churn and lets adjacent taps coalesce.
template <int CIN>
__device__ __forceinline__ v16h build_b(const float* __restrict__ w, bool hi) {
  v16h b;
  const _Float16 z = (_Float16)0.f;
#pragma unroll
  for (int e = 0; e < 16; ++e) b[e] = z;
  if (!hi) {
#pragma unroll
    for (int e = 0; e < 9; ++e) b[e] = (_Float16)w[e];          // p0 t0..8
    if (CIN == 3) {
#pragma unroll
      for (int e = 9; e < 16; ++e) b[e] = (_Float16)w[9 + e];   // p2 t0..6
    }
  } else {
#pragma unroll
    for (int e = 0; e < 9; ++e) b[e] = (_Float16)w[9 + e];      // p1 t0..8
    if (CIN == 3) {
      b[9]  = (_Float16)w[25];                                  // p2 t7
      b[10] = (_Float16)w[26];                                  // p2 t8
    }
  }
  return b;
}

// ---------------------------------------------------------------------------
// Fused conv_h (2->150) + conv_r (150->1), optional fused 2x2 maxpool on load.
// One block = one (batch, 8-row strip). A matrices are built once per wave and
// cached in registers across all 10 channel groups; h staged in LDS (stride 18
// -> 8B-aligned rows, float2 reads) then reduced into r with VALU FMAs.
// SAME-padding of h = ring-zero pass.
// ---------------------------------------------------------------------------
template <int OUT_S, bool POOL>
__global__ __launch_bounds__(256) void k_hr(const float* __restrict__ X,
                                            const float* __restrict__ Wh,
                                            const float* __restrict__ Wr,
                                            float* __restrict__ Rout) {
  constexpr int TS = 8;
  constexpr int STRIPS = OUT_S / TS;
  constexpr int XH = TS + 4, XW = OUT_S + 4;
  constexpr int XPSZ = XH * XW;
  constexpr int HH = TS + 2, HW = OUT_S + 2;
  constexpr int HP = HH * HW;
  constexpr int NT = ((HP + 127) / 128) * 8;  // tiles, multiple of 8
  constexpr int TPW = NT / 8;
  constexpr int HROWS = NT * 16;
  constexpr int PXB = TS * OUT_S;
  constexpr int PXT = (PXB + 255) / 256;

  __shared__ _Float16 Xs[2 * XPSZ];
  __shared__ float Hs[HROWS][18];
  __shared__ float Wrs[150 * 9];

  const int tid  = threadIdx.x;
  const int lane = tid & 31;
  const int wave = tid >> 5;
  const bool hi  = lane >= 16;
  const int n     = blockIdx.x / STRIPS;
  const int strip = blockIdx.x % STRIPS;
  const int row0  = strip * TS;

  for (int idx = tid; idx < 2 * XPSZ; idx += 256) {
    int c = idx / XPSZ;
    int rem = idx - c * XPSZ;
    int i = rem / XW, j = rem - i * XW;
    int gy = row0 - 2 + i, gx = j - 2;
    float v = 0.f;
    if (gy >= 0 && gy < OUT_S && gx >= 0 && gx < OUT_S) {
      if (POOL) {
        const float* xp = X + ((size_t)(n * 2 + c) * 64 + 2 * gy) * 64 + 2 * gx;
        v = fmaxf(fmaxf(xp[0], xp[1]), fmaxf(xp[64], xp[65]));
      } else {
        v = X[((size_t)(n * 2 + c) * OUT_S + gy) * OUT_S + gx];
      }
    }
    Xs[idx] = (_Float16)v;
  }
  for (int idx = tid; idx < 150 * 9; idx += 256) Wrs[idx] = Wr[idx];
  __syncthreads();

  // --- build & cache this wave's A matrices (shared by all 10 groups) ---
  v16h areg[TPW];
  int tbase[TPW];
#pragma unroll
  for (int u = 0; u < TPW; ++u) {
    const int base = (wave + u * 8) * 16;
    tbase[u] = base;
    int pos = base + (lane & 15);
    pos = (pos < HP) ? pos : (HP - 1);
    const int hr = pos / HW;
    const int hc = pos - hr * HW;
    areg[u] = build_a<2, XPSZ, XW>(Xs, hr * XW + hc, hi);
  }

  float racc[PXT];
#pragma unroll
  for (int j = 0; j < PXT; ++j) racc[j] = 0.f;

  for (int g = 0; g < 10; ++g) {
    const int ch0 = g * 16;
    const int o = ch0 + (lane & 15);
    const v16h bmat = build_b<2>(Wh + (size_t)((o < 150) ? o : 149) * 18, hi);

    const int ch = lane & 15;
    const int moff = hi ? 8 : 0;
#pragma unroll
    for (int u = 0; u < TPW; ++u) {
      v8f cacc = {};
      cacc = __builtin_amdgcn_wmma_f32_16x16x32_f16(false, areg[u], false, bmat,
                                                    (short)0, cacc, false, false);
      const int rw = tbase[u] + moff;
#pragma unroll
      for (int i = 0; i < 8; ++i) Hs[rw + i][ch] = cacc[i];
    }
    __syncthreads();
    // SAME-padding of h: zero rows/cols that fall outside the image
    for (int i = tid; i < HH * 32; i += 256)
      Hs[(i >> 5) * HW + ((i & 16) ? (HW - 1) : 0)][i & 15] = 0.f;
    if (row0 == 0)
      for (int i = tid; i < HW * 16; i += 256) Hs[i >> 4][i & 15] = 0.f;
    if (row0 + TS == OUT_S)
      for (int i = tid; i < HW * 16; i += 256) Hs[(HH - 1) * HW + (i >> 4)][i & 15] = 0.f;
    __syncthreads();

    const int omax = (150 - ch0) < 16 ? (150 - ch0) : 16;  // 16 or 6: even
#pragma unroll
    for (int j = 0; j < PXT; ++j) {
      int p = tid + 256 * j;
      if (p < PXB) {
        int py = p / OUT_S, px = p - py * OUT_S;  // OUT_S power of two
        float acc = racc[j];
        for (int dy = 0; dy < 3; ++dy)
          for (int dx = 0; dx < 3; ++dx) {
            const float2* hp2 = (const float2*)&Hs[(py + dy) * HW + (px + dx)][0];
            const float* wp = &Wrs[ch0 * 9 + dy * 3 + dx];
            for (int oo = 0; oo < omax / 2; ++oo) {
              float2 h2 = hp2[oo];
              acc = fmaf(h2.x, wp[(2 * oo) * 9], acc);
              acc = fmaf(h2.y, wp[(2 * oo + 1) * 9], acc);
            }
          }
        racc[j] = acc;
      }
    }
    __syncthreads();
  }
#pragma unroll
  for (int j = 0; j < PXT; ++j) {
    int p = tid + 256 * j;
    if (p < PXB) {
      int py = p / OUT_S, px = p - py * OUT_S;
      Rout[((size_t)n * OUT_S + row0 + py) * OUT_S + px] = racc[j];
    }
  }
}

// bilinear 2x upsample (half-pixel centers, edge clamp), 32x32 -> 64x64
__device__ __forceinline__ float upsample_at(const float* __restrict__ v, int y, int x) {
  int jy = y >> 1, jx = x >> 1;
  int y0, y1, x0, x1;
  float wy0, wx0;
  if (y & 1) { y0 = jy; y1 = (jy + 1 < 32) ? jy + 1 : 31; wy0 = 0.75f; }
  else       { y0 = (jy - 1 >= 0) ? jy - 1 : 0; y1 = jy;  wy0 = 0.25f; }
  if (x & 1) { x0 = jx; x1 = (jx + 1 < 32) ? jx + 1 : 31; wx0 = 0.75f; }
  else       { x0 = (jx - 1 >= 0) ? jx - 1 : 0; x1 = jx;  wx0 = 0.25f; }
  float wy1 = 1.f - wy0, wx1 = 1.f - wx0;
  return wy0 * (wx0 * v[y0 * 32 + x0] + wx1 * v[y0 * 32 + x1]) +
         wy1 * (wx0 * v[y1 * 32 + x0] + wx1 * v[y1 * 32 + x1]);
}

// ---------------------------------------------------------------------------
// VI recurrence: one block per batch image, planes resident in LDS, 16 steps
// with block barriers. q-conv = im2col WMMA GEMM; channel-max via __shfl_xor
// over the 16-lane halves (wave32 C layout). FINAL adds last q-conv on
// v=s/denom, BN statistics (register accum -> LDS -> global f32 atomics) and
// the (S1,S2) gather.
// ---------------------------------------------------------------------------
template <int S, int CIN, bool FINAL>
__global__ __launch_bounds__(256) void k_vi(const float* __restrict__ P0,
                                            const float* __restrict__ P1,
                                            const float* __restrict__ Wq,
                                            float* __restrict__ Vout,
                                            const int* __restrict__ S1,
                                            const int* __restrict__ S2,
                                            float* __restrict__ qsel,
                                            float* __restrict__ gstats,
                                            float denom) {
  constexpr int PW  = S + 2;
  constexpr int PSZ = PW * PW;
  constexpr int NPX = S * S;
  constexpr int NT  = NPX / 16;
  constexpr int TPW = NT / 8;
  constexpr int PXT = NPX / 256;
  constexpr int K9  = CIN * 9;

  __shared__ _Float16 PL[CIN * PSZ];   // plane-contiguous: c*PSZ (last = v)
  __shared__ float vnew[NPX];
  __shared__ float bsum[16], bsq[16];

  const int tid  = threadIdx.x;
  const int lane = tid & 31;
  const int wave = tid >> 5;
  const bool hi  = lane >= 16;
  const int n    = blockIdx.x;
  constexpr int VOFF = (CIN - 1) * PSZ;

  for (int idx = tid; idx < PSZ; idx += 256) {
    int i = idx / PW, j = idx - (idx / PW) * PW;
    int gy = i - 1, gx = j - 1;
    bool in = (gy >= 0) && (gy < S) && (gx >= 0) && (gx < S);
    float a = 0.f, b = 0.f;
    if (in) {
      if (CIN == 3) {
        a = upsample_at(P0 + (size_t)n * 1024, gy, gx);  // m = upsample2(v1)
        b = P1[(size_t)n * NPX + gy * S + gx];           // r
      } else {
        a = P0[(size_t)n * NPX + gy * S + gx];           // r1
      }
    }
    PL[idx] = (_Float16)a;
    if (CIN == 3) PL[PSZ + idx] = (_Float16)b;
    PL[VOFF + idx] = (_Float16)0.f;                      // v = 0
  }
  if (tid < 16) { bsum[tid] = 0.f; bsq[tid] = 0.f; }
  __syncthreads();

  float sacc[PXT];
#pragma unroll
  for (int j = 0; j < PXT; ++j) sacc[j] = 0.f;

  const int nn = lane & 15;
  const bool ov = nn < 10;
  const int ch = lane & 15;
  const int moff = hi ? 8 : 0;

  for (int k = 0; k < 16; ++k) {
    const v16h bmat =
        build_b<CIN>(Wq + (size_t)(k * 10 + (ov ? nn : 9)) * K9, hi);
#pragma unroll 1
    for (int u = 0; u < TPW; ++u) {
      const int base = (wave + u * 8) * 16;
      const int p = base + (lane & 15);
      const int py = p / S;   // S power of two -> shift
      const v16h amat = build_a<CIN, PSZ, PW>(PL, p + py * 2, hi);
      v8f cacc = {};
      cacc = __builtin_amdgcn_wmma_f32_16x16x32_f16(false, amat, false, bmat,
                                                    (short)0, cacc, false, false);
      float vals[8];
#pragma unroll
      for (int i = 0; i < 8; ++i) vals[i] = ov ? cacc[i] : -__builtin_inff();
#pragma unroll
      for (int off = 1; off < 16; off <<= 1)
#pragma unroll
        for (int i = 0; i < 8; ++i) vals[i] = fmaxf(vals[i], __shfl_xor(vals[i], off, 32));
      if (ch == 0) {
#pragma unroll
        for (int i = 0; i < 8; ++i) vnew[base + moff + i] = vals[i];
      }
    }
    __syncthreads();
#pragma unroll
    for (int j = 0; j < PXT; ++j) {
      int p = tid + 256 * j;
      float vn = vnew[p];
      sacc[j] += vn;
      PL[VOFF + (p / S + 1) * PW + (p & (S - 1)) + 1] = (_Float16)vn;
    }
    __syncthreads();
  }

  const float inv_denom = 1.0f / denom;
  if constexpr (!FINAL) {
#pragma unroll
    for (int j = 0; j < PXT; ++j) {
      int p = tid + 256 * j;
      Vout[(size_t)n * NPX + p] = sacc[j] * inv_denom;   // v1 = s/denom
    }
  } else {
#pragma unroll
    for (int j = 0; j < PXT; ++j) {
      int p = tid + 256 * j;
      PL[VOFF + (p / S + 1) * PW + (p & (S - 1)) + 1] = (_Float16)(sacc[j] * inv_denom);
    }
    __syncthreads();

    const int ps = S1[n] * S + S2[n];
    const v16h bmat =
        build_b<CIN>(Wq + (size_t)(15 * 10 + (ov ? nn : 9)) * K9, hi);
    float lsum = 0.f, lsq = 0.f;
#pragma unroll 1
    for (int u = 0; u < TPW; ++u) {
      const int base = (wave + u * 8) * 16;
      const int p = base + (lane & 15);
      const int py = p / S;
      const v16h amat = build_a<CIN, PSZ, PW>(PL, p + py * 2, hi);
      v8f cacc = {};
      cacc = __builtin_amdgcn_wmma_f32_16x16x32_f16(false, amat, false, bmat,
                                                    (short)0, cacc, false, false);
      if (ov) {
#pragma unroll
        for (int i = 0; i < 8; ++i) {
          float q = cacc[i];
          lsum += q;
          lsq  += q * q;
          if (base + moff + i == ps) qsel[n * 16 + ch] = q;
        }
      }
    }
    if (ov) { atomicAdd(&bsum[ch], lsum); atomicAdd(&bsq[ch], lsq); }
    __syncthreads();
    if (tid < 10) {
      atomicAdd(&gstats[tid], bsum[tid]);
      atomicAdd(&gstats[16 + tid], bsq[tid]);
    }
  }
}

// ---------------------------------------------------------------------------
__global__ void k_head(const float* __restrict__ qsel, const float* __restrict__ gstats,
                       const float* __restrict__ gamma, const float* __restrict__ beta,
                       const float* __restrict__ w1, const float* __restrict__ w2,
                       float* __restrict__ out) {
  int n = threadIdx.x;
  if (n >= 128) return;
  const float inv = 1.0f / (128.0f * 64.0f * 64.0f);
  float qn[10];
#pragma unroll
  for (int c = 0; c < 10; ++c) {
    float mean = gstats[c] * inv;
    float var  = gstats[16 + c] * inv - mean * mean;
    qn[c] = (qsel[n * 16 + c] - mean) * rsqrtf(var + 1e-5f) * gamma[c] + beta[c];
  }
  float a = 0.f;
#pragma unroll
  for (int c = 0; c < 10; ++c) a += qn[c] * w1[c];
  a = fmaxf(a, 0.f);
  float b[8], bm = 0.f;
#pragma unroll
  for (int j = 0; j < 8; ++j) {
    float s = 0.f;
#pragma unroll
    for (int c = 0; c < 10; ++c) s += qn[c] * w2[j * 10 + c];
    b[j] = fmaxf(s, 0.f);
    bm += b[j];
  }
  bm *= 0.125f;
#pragma unroll
  for (int j = 0; j < 8; ++j) out[n * 8 + j] = a + (b[j] - bm);
}

__global__ void k_zero(float* __restrict__ g) {
  if (threadIdx.x < 32) g[threadIdx.x] = 0.f;
}

// ---------------------------------------------------------------------------
extern "C" void kernel_launch(void* const* d_in, const int* in_sizes, int n_in,
                              void* d_out, int out_size, void* d_ws, size_t ws_size,
                              hipStream_t stream) {
  const float* X     = (const float*)d_in[0];
  const int*   S1    = (const int*)d_in[1];
  const int*   S2    = (const int*)d_in[2];
  const float* Wh    = (const float*)d_in[3];
  const float* Wr    = (const float*)d_in[4];
  const float* Wq    = (const float*)d_in[5];
  const float* gamma = (const float*)d_in[6];
  const float* beta  = (const float*)d_in[7];
  const float* w1    = (const float*)d_in[8];
  const float* w2    = (const float*)d_in[9];
  const float* viWh  = (const float*)d_in[10];
  const float* viWr  = (const float*)d_in[11];
  const float* viWq  = (const float*)d_in[12];

  float* ws    = (float*)d_ws;
  float* r1    = ws;            // 128*32*32
  float* v1    = ws + 131072;   // 128*32*32
  float* r     = ws + 262144;   // 128*64*64
  float* qsel  = ws + 786432;   // 128*16
  float* stats = ws + 788480;   // 32

  k_zero<<<1, 32, 0, stream>>>(stats);
  k_hr<32, true><<<128 * 4, 256, 0, stream>>>(X, viWh, viWr, r1);
  k_vi<32, 2, false><<<128, 256, 0, stream>>>(r1, nullptr, viWq, v1, nullptr,
                                              nullptr, nullptr, nullptr, 4.0f);
  k_hr<64, false><<<128 * 8, 256, 0, stream>>>(X, Wh, Wr, r);
  k_vi<64, 3, true><<<128, 256, 0, stream>>>(v1, r, Wq, nullptr, S1, S2,
                                             qsel, stats, 16.0f);
  k_head<<<1, 128, 0, stream>>>(qsel, stats, gamma, beta, w1, w2, (float*)d_out);
}
